// MoEFFN_46153718563474
// MI455X (gfx1250) — matrix operand order repaired and checked by reading, not compiled, not run on hardware
//
#include <hip/hip_runtime.h>
#include <hip/hip_bf16.h>

#define DMODEL 1024
#define MLP    4096
#define NEXP   8
#define NTOK   4096
#define TILE_M 128                 // token tile
#define TILE_N 128                 // output-column tile
#define BK     32
#define MAX_TILES 40               // <= 4096/128 + 8 pad tiles
#define CAP   (MAX_TILES * TILE_M) // 5120 padded slots
#define LDW   40                   // LDS row stride (ushorts), 16B-aligned, bank-staggered

typedef __attribute__((ext_vector_type(16))) __bf16         v16bf;
typedef __attribute__((ext_vector_type(8)))  float          v8f;
typedef __attribute__((ext_vector_type(8)))  unsigned short ushort8;
typedef __attribute__((ext_vector_type(4)))  int            v4i;

typedef __attribute__((address_space(1))) v4i glob_v4i;   // global int4
typedef __attribute__((address_space(3))) v4i lds_v4i;    // LDS int4

union Frag  { v16bf bf; ushort8 h[2]; };
union BPack { ushort8 v[2]; unsigned u[8]; };

#if defined(__has_builtin)
#if __has_builtin(__builtin_amdgcn_global_load_async_to_lds_b128)
#define HAVE_ASYNC_LDS 1
#endif
#if __has_builtin(__builtin_amdgcn_s_wait_asynccnt)
#define HAVE_WAIT_ASYNC 1
#endif
#if __has_builtin(__builtin_amdgcn_cvt_pk_bf16_f32)
#define HAVE_PK_BF16 1
#endif
#endif

#if defined(HAVE_ASYNC_LDS)
__device__ __forceinline__ void wait_async_le0() {
#if defined(HAVE_WAIT_ASYNC)
    __builtin_amdgcn_s_wait_asynccnt(0);
#else
    asm volatile("s_wait_asynccnt 0" ::: "memory");
#endif
}
#endif

// f32 -> bf16 (RNE) via VOP3P V_FMA_MIXLO_BF16 (CDNA5 ISA op 62): 1 VALU op.
__device__ __forceinline__ unsigned short f2bf(float a) {
    unsigned d;
    asm("v_fma_mixlo_bf16 %0, %1, 1.0, 0" : "=v"(d) : "v"(a));
    return (unsigned short)d;
}

// packed f32x2 -> bf16x2: mixlo writes d[15:0], mixhi writes d[31:16].
__device__ __forceinline__ unsigned pk2bf(float a, float b) {
#if defined(HAVE_PK_BF16)
    typedef __attribute__((ext_vector_type(2))) __bf16 v2bf;
    v2bf r = __builtin_amdgcn_cvt_pk_bf16_f32(a, b);
    return __builtin_bit_cast(unsigned, r);
#else
    unsigned d;
    asm("v_fma_mixlo_bf16 %0, %1, 1.0, 0\n\t"
        "v_fma_mixhi_bf16 %0, %2, 1.0, 0"
        : "=&v"(d) : "v"(a), "v"(b));
    return d;
#endif
}

// ---------------------------------------------------------------------------
// 1) Router: one wave per token. logits = x @ w_gate + b_gate, softmax top-1.
// ---------------------------------------------------------------------------
__global__ __launch_bounds__(256) void moe_router(
    const float* __restrict__ x, const float* __restrict__ wg,
    const float* __restrict__ bg, int* __restrict__ counts,
    int* __restrict__ bucket, float* __restrict__ route_p)
{
    int tok  = (blockIdx.x * blockDim.x + threadIdx.x) >> 5;
    int lane = threadIdx.x & 31;
    if (tok >= NTOK) return;
    const float* xr = x + (size_t)tok * DMODEL;
    float acc[NEXP] = {};
    for (int d = lane; d < DMODEL; d += 32) {
        float xv = xr[d];
        const float* wr = wg + (size_t)d * NEXP;
        #pragma unroll
        for (int e = 0; e < NEXP; ++e) acc[e] += xv * wr[e];
    }
    #pragma unroll
    for (int off = 16; off; off >>= 1)
        #pragma unroll
        for (int e = 0; e < NEXP; ++e) acc[e] += __shfl_xor(acc[e], off, 32);
    if (lane == 0) {
        float mx = -1e30f; int mi = 0;
        #pragma unroll
        for (int e = 0; e < NEXP; ++e) {
            acc[e] += bg[e];
            if (acc[e] > mx) { mx = acc[e]; mi = e; }
        }
        float s = 0.f;
        #pragma unroll
        for (int e = 0; e < NEXP; ++e) s += __expf(acc[e] - mx);
        int pos = atomicAdd(&counts[mi], 1);
        bucket[mi * NTOK + pos] = tok;
        route_p[tok] = 1.0f / s;              // top-1 prob = exp(0)/sum
    }
}

// ---------------------------------------------------------------------------
// 2) Scan: pad each expert's region to 128-token tiles, tile->expert map.
// ---------------------------------------------------------------------------
__global__ void moe_scan(const int* __restrict__ counts,
                         int* __restrict__ expert_off,
                         int* __restrict__ tile_expert)
{
    if (threadIdx.x != 0) return;
    int off = 0, t = 0;
    for (int e = 0; e < NEXP; ++e) {
        expert_off[e] = off;
        int nt = (counts[e] + TILE_M - 1) / TILE_M;
        for (int i = 0; i < nt; ++i) tile_expert[t++] = e;
        off += nt * TILE_M;
    }
    for (; t < MAX_TILES; ++t) tile_expert[t] = -1;
}

// ---------------------------------------------------------------------------
// 3) Gather: expert-contiguous bf16 activation rows; zero rows for pad slots.
// ---------------------------------------------------------------------------
__global__ __launch_bounds__(256) void moe_gather(
    const float* __restrict__ x, const int* __restrict__ counts,
    const int* __restrict__ expert_off, const int* __restrict__ tile_expert,
    const int* __restrict__ bucket, const float* __restrict__ route_p,
    unsigned short* __restrict__ xg, int* __restrict__ slot_token,
    float* __restrict__ slot_scale)
{
    int slot = blockIdx.x;
    int e = tile_expert[slot / TILE_M];
    if (e < 0) return;
    int local = slot - expert_off[e];
    int tok = (local < counts[e]) ? bucket[e * NTOK + local] : -1;
    if (threadIdx.x == 0) {
        slot_token[slot] = tok;
        slot_scale[slot] = (tok >= 0) ? route_p[tok] : 0.f;
    }
    unsigned short* dst = xg + (size_t)slot * DMODEL;
    int i = threadIdx.x * 4;                  // 256 thr * 4 = 1024 = DMODEL
    if (tok >= 0) {
        const float* src = x + (size_t)tok * DMODEL;
        float4 v = *(const float4*)(src + i);
        *(unsigned*)(dst + i)     = pk2bf(v.x, v.y);
        *(unsigned*)(dst + i + 2) = pk2bf(v.z, v.w);
    } else {
        *(uint2*)(dst + i) = make_uint2(0u, 0u);
    }
}

// ---------------------------------------------------------------------------
// 4) Fused expert GEMM, 128x128 tile, BK=32, double-buffered LDS.
//    8 waves, each owns a 32x64 sub-tile -> 8 v_wmma per wave per K-step.
//    A (bf16 activations) moved global->LDS via async-to-LDS when available;
//    W (f32 weights) converted to bf16 (v_fma_mixlo/hi_bf16) on the load path.
// ---------------------------------------------------------------------------
template<int KDIM, int NCOLS, bool SECOND>
__global__ __launch_bounds__(256) void moe_gemm(
    const unsigned short* __restrict__ Abuf,
    const float* __restrict__ Wall,
    const float* __restrict__ bias,
    unsigned short* __restrict__ Hout,
    float* __restrict__ Out,
    const int* __restrict__ tile_expert,
    const int* __restrict__ slot_token,
    const float* __restrict__ slot_scale)
{
    __shared__ unsigned short sA[2][TILE_M * LDW];  // [row][k]
    __shared__ unsigned short sB[2][TILE_N * LDW];  // [col][k]  (B^T)

    int e = tile_expert[blockIdx.y];
    if (e < 0) return;                        // uniform exit before barriers
    const int slot0 = blockIdx.y * TILE_M;
    const int n0    = blockIdx.x * TILE_N;
    const unsigned short* A  = Abuf + (size_t)slot0 * KDIM;
    const float*          Wp = Wall + (size_t)e * KDIM * NCOLS + n0;
    const float*          bv = bias + (size_t)e * NCOLS + n0;

    const int tid  = threadIdx.x;
    const int lane = tid & 31;
    const int wv   = tid >> 5;
    const int wr   = (wv & 3) * 32;           // wave row base (4 groups x 32)
    const int wc   = (wv >> 2) * 64;          // wave col base (2 groups x 64)
    const int half = lane >> 4;
    const int l    = lane & 15;

    // loader mapping
    const int arow = tid >> 1,  ac  = (tid & 1) << 4;   // A: 128x32 bf16
    const int bcol = tid & 127, bkh = (tid >> 7) << 4;  // B: col-per-thread

    float rb[16];
    auto loadB = [&](int k0) {
        #pragma unroll
        for (int kk = 0; kk < 16; ++kk)
            rb[kk] = Wp[(size_t)(k0 + bkh + kk) * NCOLS + bcol];
    };
    auto storeB = [&](int buf) {
        BPack up;
        #pragma unroll
        for (int kk = 0; kk < 8; ++kk)
            up.u[kk] = pk2bf(rb[2 * kk], rb[2 * kk + 1]);
        *(ushort8*)&sB[buf][bcol * LDW + bkh]     = up.v[0];
        *(ushort8*)&sB[buf][bcol * LDW + bkh + 8] = up.v[1];
    };
    auto issueA = [&](int k0, int buf) {
        const unsigned short* g = A + (size_t)arow * KDIM + k0 + ac;
        unsigned short* s = &sA[buf][arow * LDW + ac];
#if defined(HAVE_ASYNC_LDS)
        __builtin_amdgcn_global_load_async_to_lds_b128(
            (glob_v4i*)g, (lds_v4i*)s, 0, 0);
        __builtin_amdgcn_global_load_async_to_lds_b128(
            (glob_v4i*)(g + 8), (lds_v4i*)(s + 8), 0, 0);
#else
        ushort8 a0 = *(const ushort8*)g;
        ushort8 a1 = *(const ushort8*)(g + 8);
        *(ushort8*)s       = a0;
        *(ushort8*)(s + 8) = a1;
#endif
    };

    v8f acc[2][4];
    #pragma unroll
    for (int i = 0; i < 2; ++i)
        #pragma unroll
        for (int j = 0; j < 4; ++j) acc[i][j] = (v8f){};

    // prologue: stage tile 0 into buffer 0
    loadB(0);
    issueA(0, 0);
    storeB(0);
#if defined(HAVE_ASYNC_LDS)
    wait_async_le0();
#endif
    __syncthreads();

    int p = 0;
    for (int k0 = 0; k0 < KDIM; k0 += BK) {
        const bool has_next = (k0 + BK) < KDIM;
        if (has_next) {
            loadB(k0 + BK);                  // global f32 weights -> regs
            issueA(k0 + BK, p ^ 1);          // bf16 activations -> LDS (async)
            if (k0 + 2 * BK < KDIM)          // prefetch weight stream ahead
                __builtin_prefetch(&Wp[(size_t)(k0 + 2 * BK + bkh) * NCOLS + bcol], 0, 1);
        }

        // fragments: element s of lane half h holds k = 16h + s in both A and B
        const unsigned short* sAp = sA[p];
        const unsigned short* sBp = sB[p];
        Frag fa[2], fb[4];
        #pragma unroll
        for (int i = 0; i < 2; ++i) {
            const unsigned short* pa = &sAp[(wr + i * 16 + l) * LDW + half * 16];
            fa[i].h[0] = *(const ushort8*)pa;
            fa[i].h[1] = *(const ushort8*)(pa + 8);
        }
        #pragma unroll
        for (int j = 0; j < 4; ++j) {
            const unsigned short* pb = &sBp[(wc + j * 16 + l) * LDW + half * 16];
            fb[j].h[0] = *(const ushort8*)pb;
            fb[j].h[1] = *(const ushort8*)(pb + 8);
        }
        #pragma unroll
        for (int i = 0; i < 2; ++i)
            #pragma unroll
            for (int j = 0; j < 4; ++j)
                acc[i][j] = __builtin_amdgcn_wmma_f32_16x16x32_bf16(
                    false, fa[i].bf, false, fb[j].bf, (short)0, acc[i][j],
                    false, false);

        if (has_next) {
            storeB(p ^ 1);                   // fma_mix cvt + 2x ds_store_b128
        }
#if defined(HAVE_ASYNC_LDS)
        // Async loads complete in order; only the pair issued THIS iteration
        // (for tile k+1) is outstanding here, and it had the whole compute
        // phase to land. Must be 0 before the barrier so next iteration's
        // fragment reads see the data.
        wait_async_le0();
#endif
        __syncthreads();
        p ^= 1;
    }

    // C/D layout: lane<16 -> M=r, N=l ; lane>=16 -> M=8+r, N=l
    #pragma unroll
    for (int i = 0; i < 2; ++i) {
        const int gm0 = slot0 + wr + i * 16 + half * 8;
        #pragma unroll
        for (int j = 0; j < 4; ++j) {
            int cn = wc + j * 16 + l;
            int gn = n0 + cn;
            float bval = bv[cn];
            #pragma unroll
            for (int r = 0; r < 8; ++r) {
                float v = acc[i][j][r] + bval;
                if (!SECOND) {
                    v = fmaxf(v, 0.f);
                    Hout[(size_t)(gm0 + r) * MLP + gn] = f2bf(v);
                } else {
                    int slot = gm0 + r;
                    int tok = slot_token[slot];
                    if (tok >= 0)
                        Out[(size_t)tok * DMODEL + gn] = v * slot_scale[slot];
                }
            }
        }
    }
}

// ---------------------------------------------------------------------------
extern "C" void kernel_launch(void* const* d_in, const int* in_sizes, int n_in,
                              void* d_out, int out_size, void* d_ws, size_t ws_size,
                              hipStream_t stream)
{
    (void)in_sizes; (void)n_in; (void)out_size; (void)ws_size;
    const float* x  = (const float*)d_in[0];
    const float* wg = (const float*)d_in[1];
    const float* bg = (const float*)d_in[2];
    const float* W1 = (const float*)d_in[3];
    const float* b1 = (const float*)d_in[4];
    const float* W2 = (const float*)d_in[5];
    const float* b2 = (const float*)d_in[6];
    float* out = (float*)d_out;

    char* ws = (char*)d_ws;
    size_t o = 0;
    auto alloc = [&](size_t bytes) -> void* {
        void* p = ws + o; o += (bytes + 255) & ~(size_t)255; return p;
    };
    int*   counts      = (int*)  alloc(NEXP * 4);
    int*   expert_off  = (int*)  alloc(NEXP * 4);
    int*   tile_expert = (int*)  alloc(MAX_TILES * 4);
    int*   bucket      = (int*)  alloc((size_t)NEXP * NTOK * 4);
    int*   slot_token  = (int*)  alloc((size_t)CAP * 4);
    float* route_p     = (float*)alloc((size_t)NTOK * 4);
    float* slot_scale  = (float*)alloc((size_t)CAP * 4);
    unsigned short* xg = (unsigned short*)alloc((size_t)CAP * DMODEL * 2);
    unsigned short* hb = (unsigned short*)alloc((size_t)CAP * MLP * 2);

    (void)hipMemsetAsync(counts, 0, NEXP * 4, stream);
    moe_router<<<NTOK / 8, 256, 0, stream>>>(x, wg, bg, counts, bucket, route_p);
    moe_scan<<<1, 32, 0, stream>>>(counts, expert_off, tile_expert);
    moe_gather<<<CAP, 256, 0, stream>>>(x, counts, expert_off, tile_expert,
                                        bucket, route_p, xg, slot_token, slot_scale);
    moe_gemm<DMODEL, MLP, false><<<dim3(MLP / TILE_N, MAX_TILES), 256, 0, stream>>>(
        xg, W1, b1, hb, nullptr, tile_expert, slot_token, slot_scale);
    moe_gemm<MLP, DMODEL, true><<<dim3(DMODEL / TILE_N, MAX_TILES), 256, 0, stream>>>(
        hb, W2, b2, nullptr, out, tile_expert, slot_token, slot_scale);
}
